// NeuronCircuit_52819507806523
// MI455X (gfx1250) — matrix operand order repaired and checked by reading, not compiled, not run on hardware
//
#include <hip/hip_runtime.h>
#include <hip/hip_fp16.h>

typedef __attribute__((ext_vector_type(16))) _Float16     v16h;
typedef __attribute__((ext_vector_type(8)))  float        v8f;
typedef __attribute__((ext_vector_type(4)))  unsigned int u32x4;
typedef __attribute__((ext_vector_type(8)))  int          i32x8;
typedef __attribute__((ext_vector_type(4)))  int          i32x4;

#define B_   4
#define S_   2048
#define D_   1024
#define R_   512
#define H_   8
#define NC_  8
#define NE_  8
#define DH_  64
#define BS_  (B_ * S_)

// LDS byte offset of a __shared__ object (flat address truncates to LDS offset).
__device__ __forceinline__ unsigned int lds_off(const void* p) {
    return (unsigned int)(unsigned long long)p;
}

// Async global->LDS 16-byte copy (per-lane), tracked by ASYNCcnt.
__device__ __forceinline__ void async_b128(unsigned int ldst, const void* g) {
    unsigned long long ga = (unsigned long long)g;
    asm volatile("global_load_async_to_lds_b128 %0, %1, off"
                 :: "v"(ldst), "v"(ga) : "memory");
}

__device__ __forceinline__ void wait_async0() {
    asm volatile("s_wait_asynccnt 0" ::: "memory");
}

// ---------------------------------------------------------------------------
// fp32 -> f16 bulk convert (for TDM/async-copyable weight images).
// ---------------------------------------------------------------------------
__global__ __launch_bounds__(256) void f32_to_f16_kernel(
    const float* __restrict__ src, _Float16* __restrict__ dst, int n)
{
    int i = (blockIdx.x * 256 + threadIdx.x) * 4;
    if (i < n) {
        float4 v = *(const float4*)(src + i);
        dst[i + 0] = (_Float16)v.x;
        dst[i + 1] = (_Float16)v.y;
        dst[i + 2] = (_Float16)v.z;
        dst[i + 3] = (_Float16)v.w;
    }
}

// ---------------------------------------------------------------------------
// Router: one wave per token. logits[n] = x_row . w_router[n], softmax over 8.
// ---------------------------------------------------------------------------
__global__ __launch_bounds__(256) void router_kernel(
    const float* __restrict__ x, const float* __restrict__ wr,
    float* __restrict__ wout, int Dlen)
{
    const int lane = threadIdx.x & 31;
    const int wave = threadIdx.x >> 5;
    const int token = blockIdx.x * 8 + wave;
    const float* xrow = x + (size_t)token * Dlen;

    float acc[8] = {0.f, 0.f, 0.f, 0.f, 0.f, 0.f, 0.f, 0.f};
    for (int d = lane; d < Dlen; d += 32) {
        float xv = xrow[d];
#pragma unroll
        for (int n = 0; n < 8; ++n)
            acc[n] = fmaf(xv, wr[n * Dlen + d], acc[n]);
    }
#pragma unroll
    for (int n = 0; n < 8; ++n) {
#pragma unroll
        for (int off = 16; off > 0; off >>= 1)
            acc[n] += __shfl_xor(acc[n], off, 32);
    }
    if (lane == 0) {
        float mx = acc[0];
#pragma unroll
        for (int n = 1; n < 8; ++n) mx = fmaxf(mx, acc[n]);
        float e[8]; float sum = 0.f;
#pragma unroll
        for (int n = 0; n < 8; ++n) { e[n] = __expf(acc[n] - mx); sum += e[n]; }
        float inv = 1.0f / sum;
#pragma unroll
        for (int n = 0; n < 8; ++n) wout[(size_t)token * 8 + n] = e[n] * inv;
    }
}

// ---------------------------------------------------------------------------
// Fused q/k/v compress GEMM. M=BS, N=R, K=NC*D. A[t,k]=w[t,n(k)]*x[t,d(k)].
// Block: 256 threads = 8 waves (4M x 2N), tile 64x32. Grid (R/32, BS/64).
// Tiles staged via global_load_async_to_lds_b128 (ASYNCcnt).
// ---------------------------------------------------------------------------
__global__ __launch_bounds__(256) void compress_gemm_kernel(
    const float* __restrict__ x,        // [BS, D]
    const _Float16* __restrict__ cnh,   // [NC, D, R] f16
    const float* __restrict__ wq, const float* __restrict__ wk,
    const float* __restrict__ wv,
    _Float16* __restrict__ qh, _Float16* __restrict__ kh,
    _Float16* __restrict__ vh)
{
    __shared__ float    lx[64][36];   // 144B row stride (16B aligned)
    __shared__ _Float16 lb[32][40];   // 80B row stride (16B aligned)

    const int tid  = threadIdx.x;
    const int lane = tid & 31;
    const int wave = tid >> 5;
    const int wm = wave & 3, wn = wave >> 2;
    const int m0 = blockIdx.y * 64;
    const int n0 = blockIdx.x * 32;

    v8f accq = {}; v8f acck = {}; v8f accv = {};

    const int arow  = lane & 15;
    const int abase = (lane >> 4) ? 8 : 0;
    const int mrow  = m0 + wm * 16 + arow;
    float sq = 0.f, sk = 0.f, sv = 0.f;
    int curn = -1;

    for (int kk = 0; kk < NC_ * D_; kk += 32) {
        const int neuron = kk >> 10;
        const int d0     = kk & (D_ - 1);
        __syncthreads();
        // x tile: 64 rows x 32 f32 = 512 x 16B chunks, 2 per thread
#pragma unroll
        for (int rep = 0; rep < 2; ++rep) {
            int i = tid + rep * 256;
            int r = i >> 3, c4 = i & 7;
            async_b128(lds_off(&lx[r][c4 * 4]),
                       x + (size_t)(m0 + r) * D_ + d0 + c4 * 4);
        }
        // B tile: 32 rows x 32 f16 = 128 x 16B chunks
        if (tid < 128) {
            int r = tid >> 2, c8 = tid & 3;
            async_b128(lds_off(&lb[r][c8 * 8]),
                       cnh + ((size_t)neuron * D_ + d0 + r) * R_ + n0 + c8 * 8);
        }
        wait_async0();
        __syncthreads();

        if (neuron != curn) {
            curn = neuron;
            sq = wq[(size_t)mrow * 8 + neuron];
            sk = wk[(size_t)mrow * 8 + neuron];
            sv = wv[(size_t)mrow * 8 + neuron];
        }

        float af[16];
#pragma unroll
        for (int e = 0; e < 8; ++e) {
            af[e]     = lx[wm * 16 + arow][abase + e];
            af[8 + e] = lx[wm * 16 + arow][abase + 16 + e];
        }
        v16h bf;
#pragma unroll
        for (int e = 0; e < 16; ++e) bf[e] = lb[(e < 8) ? abase + e : abase + 8 + e][wn * 16 + arow];

        v16h aq, ak, av;
#pragma unroll
        for (int e = 0; e < 16; ++e) {
            aq[e] = (_Float16)(af[e] * sq);
            ak[e] = (_Float16)(af[e] * sk);
            av[e] = (_Float16)(af[e] * sv);
        }
        accq = __builtin_amdgcn_wmma_f32_16x16x32_f16(false, aq, false, bf,
                                                      (short)0, accq, false, false);
        acck = __builtin_amdgcn_wmma_f32_16x16x32_f16(false, ak, false, bf,
                                                      (short)0, acck, false, false);
        accv = __builtin_amdgcn_wmma_f32_16x16x32_f16(false, av, false, bf,
                                                      (short)0, accv, false, false);
    }

    const int ncol   = n0 + wn * 16 + (lane & 15);
    const int h      = ncol >> 6, dh = ncol & 63;
    const int rowadd = (lane >> 4) * 8;
#pragma unroll
    for (int i = 0; i < 8; ++i) {
        int m = m0 + wm * 16 + rowadd + i;
        int b = m >> 11, s = m & (S_ - 1);
        size_t o = (((size_t)b * H_ + h) * S_ + s) * DH_ + dh;
        qh[o] = (_Float16)(accq[i] * 0.125f);   // 1/sqrt(64)
        kh[o] = (_Float16)acck[i];
        vh[o] = (_Float16)accv[i];
    }
}

// ---------------------------------------------------------------------------
// Causal flash attention. Grid (S/64, B*H), block 128 (4 waves).
// K/V 64-key chunks (8KB contiguous f16) fetched with the Tensor Data Mover.
// ---------------------------------------------------------------------------
__global__ __launch_bounds__(128) void attn_kernel(
    const _Float16* __restrict__ qh, const _Float16* __restrict__ kh,
    const _Float16* __restrict__ vh, float* __restrict__ aout)
{
    __shared__ _Float16 lk[64][DH_];
    __shared__ _Float16 lv[64][DH_];
    __shared__ _Float16 lp[4][16][36];   // per-wave P staging (C->A transpose)

    const int tid = threadIdx.x, lane = tid & 31, wave = tid >> 5;
    const int qblk = blockIdx.x;
    const int bh   = blockIdx.y;
    const int b = bh >> 3, h = bh & 7;
    const int qbase = qblk * 64;

    const _Float16* qb = qh + (size_t)bh * S_ * DH_;
    const _Float16* kb = kh + (size_t)bh * S_ * DH_;
    const _Float16* vb = vh + (size_t)bh * S_ * DH_;

    const int arow  = lane & 15;
    const int abase = (lane >> 4) ? 8 : 0;
    const int crow  = (lane >> 4) * 8;
    const int ccol  = lane & 15;

    const unsigned int lkOff = lds_off(&lk[0][0]);
    const unsigned int lvOff = lds_off(&lv[0][0]);

    // Q as two A-fragments (K = DH split 2x32), loaded once.
    const int qrow_frag = qbase + wave * 16 + arow;
    v16h qf[2];
#pragma unroll
    for (int c = 0; c < 2; ++c) {
#pragma unroll
        for (int e = 0; e < 8; ++e) {
            qf[c][e]     = qb[(size_t)qrow_frag * DH_ + c * 32 + abase + e];
            qf[c][8 + e] = qb[(size_t)qrow_frag * DH_ + c * 32 + abase + 16 + e];
        }
    }

    v8f acco[4] = {{}, {}, {}, {}};
    float rm[8], rl[8];
#pragma unroll
    for (int i = 0; i < 8; ++i) { rm[i] = -1e30f; rl[i] = 0.f; }

    const int qrow0 = qbase + wave * 16 + crow;
    const int nchunks = qblk + 1;

    for (int c = 0; c < nchunks; ++c) {
        const int kbase = c * 64;
        __syncthreads();
        if (wave == 0) {
            // TDM: 1-D copy of 4096 f16 (8KB) per descriptor; D# per ISA 8.3/8.4.
            unsigned long long gaK =
                (unsigned long long)(kb + (size_t)kbase * DH_);
            unsigned long long gaV =
                (unsigned long long)(vb + (size_t)kbase * DH_);
            // group1: data_size=2B; dim0=tile0=4096; dim1=tile1=1; stride0=4096
            i32x8 g1 = { (int)0x00010000, (int)(4096u << 16), (int)0x00010000,
                         (int)(4096u << 16), 1, 4096, 0, 0 };
            i32x4 z4 = { 0, 0, 0, 0 };
            i32x8 z8 = { 0, 0, 0, 0, 0, 0, 0, 0 };
            u32x4 g0k = { 1u, lkOff, (unsigned int)gaK,
                          ((unsigned int)(gaK >> 32) & 0x01FFFFFFu) | 0x80000000u };
            u32x4 g0v = { 1u, lvOff, (unsigned int)gaV,
                          ((unsigned int)(gaV >> 32) & 0x01FFFFFFu) | 0x80000000u };
            __builtin_amdgcn_tensor_load_to_lds(g0k, g1, z4, z4, z8, 0);
            __builtin_amdgcn_tensor_load_to_lds(g0v, g1, z4, z4, z8, 0);
            __builtin_amdgcn_s_wait_tensorcnt(0);
        }
        __syncthreads();

        for (int jp = 0; jp < 2; ++jp) {
#pragma unroll
            for (int jj = 0; jj < 2; ++jj) {
                const int j = jp * 2 + jj;
                v8f accs = {};
#pragma unroll
                for (int kc = 0; kc < 2; ++kc) {
                    v16h bf;  // B layout: lane%16 = key col, elems = dh (table order)
#pragma unroll
                    for (int e = 0; e < 16; ++e)
                        bf[e] = lk[j * 16 + ccol][kc * 32 + ((e < 8) ? abase + e : abase + 8 + e)];
                    accs = __builtin_amdgcn_wmma_f32_16x16x32_f16(
                        false, qf[kc], false, bf, (short)0, accs, false, false);
                }
                const int keyg = kbase + j * 16 + ccol;
#pragma unroll
                for (int i = 0; i < 8; ++i) {
                    float s  = (keyg <= qrow0 + i) ? accs[i] : -1e30f;
                    float mx = s;
                    mx = fmaxf(mx, __shfl_xor(mx, 1, 32));
                    mx = fmaxf(mx, __shfl_xor(mx, 2, 32));
                    mx = fmaxf(mx, __shfl_xor(mx, 4, 32));
                    mx = fmaxf(mx, __shfl_xor(mx, 8, 32));
                    float nm = fmaxf(rm[i], mx);
                    float sc = __expf(rm[i] - nm);
                    float p  = __expf(s - nm);
                    float ps = p;
                    ps += __shfl_xor(ps, 1, 32);
                    ps += __shfl_xor(ps, 2, 32);
                    ps += __shfl_xor(ps, 4, 32);
                    ps += __shfl_xor(ps, 8, 32);
                    rl[i] = rl[i] * sc + ps;
                    rm[i] = nm;
#pragma unroll
                    for (int t = 0; t < 4; ++t) acco[t][i] *= sc;
                    lp[wave][crow + i][jj * 16 + ccol] = (_Float16)p;
                }
            }
            asm volatile("s_wait_dscnt 0" ::: "memory");
            __builtin_amdgcn_wave_barrier();

            v16h pf;  // P as A fragment, K = 32 keys
#pragma unroll
            for (int e = 0; e < 8; ++e) {
                pf[e]     = lp[wave][arow][abase + e];
                pf[8 + e] = lp[wave][arow][abase + 16 + e];
            }
#pragma unroll
            for (int t = 0; t < 4; ++t) {
                v16h vf;  // B layout: lane%16 = dh col, elems = keys (table order)
#pragma unroll
                for (int e = 0; e < 16; ++e)
                    vf[e] = lv[jp * 32 + ((e < 8) ? abase + e : abase + 8 + e)][t * 16 + ccol];
                acco[t] = __builtin_amdgcn_wmma_f32_16x16x32_f16(
                    false, pf, false, vf, (short)0, acco[t], false, false);
            }
        }
    }

#pragma unroll
    for (int i = 0; i < 8; ++i) {
        const int m = qrow0 + i;
        const float inv = 1.0f / rl[i];
#pragma unroll
        for (int t = 0; t < 4; ++t) {
            aout[((size_t)b * S_ + m) * R_ + h * 64 + t * 16 + ccol] =
                acco[t][i] * inv;
        }
    }
}

// ---------------------------------------------------------------------------
// Expand GEMM. M=BS, N=D, K=NE*R. A[t,k]=wo[t,n(k)]*attn_out[t,r(k)].
// Block 256 = 8 waves (4M x 2N), tile 64x32. Grid (D/32, BS/64). fp32 out.
// ---------------------------------------------------------------------------
__global__ __launch_bounds__(256) void expand_gemm_kernel(
    const float* __restrict__ a,        // [BS, R]
    const _Float16* __restrict__ enh,   // [NE, R, D] f16
    const float* __restrict__ wo,       // [BS, 8]
    float* __restrict__ out)            // [BS, D]
{
    __shared__ float    lx[64][36];
    __shared__ _Float16 lb[32][40];

    const int tid  = threadIdx.x;
    const int lane = tid & 31;
    const int wave = tid >> 5;
    const int wm = wave & 3, wn = wave >> 2;
    const int m0 = blockIdx.y * 64;
    const int n0 = blockIdx.x * 32;

    v8f acc = {};
    const int arow  = lane & 15;
    const int abase = (lane >> 4) ? 8 : 0;
    const int mrow  = m0 + wm * 16 + arow;
    float sc = 0.f;
    int curn = -1;

    for (int kk = 0; kk < NE_ * R_; kk += 32) {
        const int neuron = kk >> 9;
        const int r0     = kk & (R_ - 1);
        __syncthreads();
#pragma unroll
        for (int rep = 0; rep < 2; ++rep) {
            int i = tid + rep * 256;
            int r = i >> 3, c4 = i & 7;
            async_b128(lds_off(&lx[r][c4 * 4]),
                       a + (size_t)(m0 + r) * R_ + r0 + c4 * 4);
        }
        if (tid < 128) {
            int r = tid >> 2, c8 = tid & 3;
            async_b128(lds_off(&lb[r][c8 * 8]),
                       enh + ((size_t)neuron * R_ + r0 + r) * D_ + n0 + c8 * 8);
        }
        wait_async0();
        __syncthreads();

        if (neuron != curn) { curn = neuron; sc = wo[(size_t)mrow * 8 + neuron]; }

        v16h av, bf;
#pragma unroll
        for (int e = 0; e < 8; ++e) {
            av[e]     = (_Float16)(lx[wm * 16 + arow][abase + e] * sc);
            av[8 + e] = (_Float16)(lx[wm * 16 + arow][abase + 16 + e] * sc);
        }
#pragma unroll
        for (int e = 0; e < 16; ++e) bf[e] = lb[(e < 8) ? abase + e : abase + 8 + e][wn * 16 + arow];

        acc = __builtin_amdgcn_wmma_f32_16x16x32_f16(false, av, false, bf,
                                                     (short)0, acc, false, false);
    }

    const int ncol   = n0 + wn * 16 + (lane & 15);
    const int rowadd = (lane >> 4) * 8;
#pragma unroll
    for (int i = 0; i < 8; ++i) {
        int m = m0 + wm * 16 + rowadd + i;
        out[(size_t)m * D_ + ncol] = acc[i];
    }
}

// ---------------------------------------------------------------------------
extern "C" void kernel_launch(void* const* d_in, const int* in_sizes, int n_in,
                              void* d_out, int out_size, void* d_ws, size_t ws_size,
                              hipStream_t stream) {
    (void)in_sizes; (void)n_in; (void)out_size; (void)ws_size;
    const float* x   = (const float*)d_in[0];
    // d_in[1] = mask (causal, handled analytically)
    const float* cn  = (const float*)d_in[2];
    const float* en  = (const float*)d_in[3];
    const float* wrq = (const float*)d_in[4];
    const float* wrk = (const float*)d_in[5];
    const float* wrv = (const float*)d_in[6];
    const float* wro = (const float*)d_in[7];
    float* out = (float*)d_out;

    // workspace layout
    float* wq = (float*)d_ws;                          // [BS,8]
    float* wk = wq + (size_t)BS_ * 8;
    float* wv = wk + (size_t)BS_ * 8;
    float* wo = wv + (size_t)BS_ * 8;
    _Float16* qh = (_Float16*)(wo + (size_t)BS_ * 8);  // [B,H,S,DH] f16
    _Float16* kh = qh + (size_t)BS_ * R_;
    _Float16* vh = kh + (size_t)BS_ * R_;
    float* aout = (float*)(vh + (size_t)BS_ * R_);     // [BS, R] f32
    _Float16* cnh = (_Float16*)(aout + (size_t)BS_ * R_);  // [NC,D,R] f16
    _Float16* enh = cnh + (size_t)NC_ * D_ * R_;           // [NE,R,D] f16

    f32_to_f16_kernel<<<(NC_ * D_ * R_) / 1024, 256, 0, stream>>>(
        cn, cnh, NC_ * D_ * R_);
    f32_to_f16_kernel<<<(NE_ * R_ * D_) / 1024, 256, 0, stream>>>(
        en, enh, NE_ * R_ * D_);

    router_kernel<<<BS_ / 8, 256, 0, stream>>>(x, wrq, wq, D_);
    router_kernel<<<BS_ / 8, 256, 0, stream>>>(x, wrk, wk, D_);
    router_kernel<<<BS_ / 8, 256, 0, stream>>>(x, wrv, wv, D_);

    compress_gemm_kernel<<<dim3(R_ / 32, BS_ / 64), 256, 0, stream>>>(
        x, cnh, wq, wk, wv, qh, kh, vh);

    attn_kernel<<<dim3(S_ / 64, B_ * H_), 128, 0, stream>>>(qh, kh, vh, aout);

    router_kernel<<<BS_ / 8, 256, 0, stream>>>(aout, wro, wo, R_);

    expand_gemm_kernel<<<dim3(D_ / 32, BS_ / 64), 256, 0, stream>>>(
        aout, enh, wo, out);
}
